// LSTMAttentionLayer_60327110640458
// MI455X (gfx1250) — matrix operand order repaired and checked by reading, not compiled, not run on hardware
//
#include <hip/hip_runtime.h>
#include <math.h>

typedef __attribute__((ext_vector_type(16))) __bf16 v16bf;
typedef __attribute__((ext_vector_type(8)))  float  v8f;

#define B_ 64
#define T_ 2048
#define D_ 256
#define U_ 256

// ws layout (bytes):
//   [0,       65536)  : w1q   float[B][U]
//   [65536,  196608)  : packB bf16  [16 nb][8 kb][32 lanes][16 elems]
//   [196608, 720896)  : scores float[B][T]

// Hardware tanh (CDNA5 V_TANH_F32 trans op) — avoids branchy libm expansion.
__device__ __forceinline__ float fast_tanh(float x) {
#if __has_builtin(__builtin_amdgcn_tanhf)
  return __builtin_amdgcn_tanhf(x);
#else
  float r;
  asm volatile("v_tanh_f32 %0, %1" : "=v"(r) : "v"(x));
  return r;
#endif
}

// ---------------------------------------------------------------------------
// Kernel 0a: w1q[b][u] = sum_d query[b][d] * W1[d][u]   (64x256, trivial)
// ---------------------------------------------------------------------------
__global__ __launch_bounds__(256) void w1q_kernel(const float* __restrict__ query,
                                                  const float* __restrict__ W1,
                                                  float* __restrict__ w1q) {
  const int b = blockIdx.x, u = threadIdx.x;
  float acc = 0.f;
#pragma unroll 8
  for (int d = 0; d < D_; ++d)
    acc = fmaf(query[b * D_ + d], W1[d * U_ + u], acc);
  w1q[b * U_ + u] = acc;
}

// ---------------------------------------------------------------------------
// Kernel 0b: pre-pack W2 (f32 -> bf16) into WMMA B-operand lane layout.
// B matrix 32x16 (KxN), 16-bit: N = lane & 15. Element j of the lane's
// v16bf -> K = j + 8*(j>>3) + 8*kh (same order as the A operand).
// ---------------------------------------------------------------------------
__global__ void packB_kernel(const float* __restrict__ W2,
                             __bf16* __restrict__ out) {
  const int nb = blockIdx.x >> 3, kb = blockIdx.x & 7;
  const int l = threadIdx.x;
  const int kh = l >> 4, nn = l & 15;
#pragma unroll
  for (int j = 0; j < 16; ++j) {
    const int k = kb * 32 + 8 * kh + j + 8 * (j >> 3);
    out[(size_t)blockIdx.x * 512 + l * 16 + j] =
        (__bf16)W2[k * U_ + nb * 16 + nn];
  }
}

// ---------------------------------------------------------------------------
// Kernel 1: scores[b][t] = sum_u scale[u]*tanh(w1q[b][u] + (value@W2)[b][t][u])
// One wave per 16-row T-tile; 8 waves/block. packB (128 KB) is staged once
// per block into LDS via async global->LDS loads, shared by all 8 waves.
// ---------------------------------------------------------------------------
__global__ __launch_bounds__(256) void scores_kernel(
    const float* __restrict__ value, const float* __restrict__ w1q,
    const float* __restrict__ scale, const __bf16* __restrict__ packB,
    float* __restrict__ scores) {
  __shared__ __align__(16) __bf16 sB[16 * 8 * 512];  // 128 KB

  const int b    = blockIdx.x;
  const int lane = threadIdx.x & 31;
  const int wave = threadIdx.x >> 5;
  const int t0   = (blockIdx.y * 8 + wave) * 16;
  const int m    = lane & 15;      // row within tile (A) / column N (B,C)
  const int kh   = lane >> 4;      // K-half selector

  // Issue async copy of packB into LDS: 256 threads x 512 B (32 x b128 each).
  // INST_OFFSET applies to both the global and LDS address.
  {
    const char* gp = (const char*)packB + (size_t)threadIdx.x * 512;
    const unsigned lp =
        (unsigned)(uintptr_t)((char*)sB + (size_t)threadIdx.x * 512);
#pragma unroll
    for (int i = 0; i < 32; ++i)
      asm volatile("global_load_async_to_lds_b128 %0, %1, off offset:%2"
                   :: "v"(lp), "v"(gp), "i"(i * 16) : "memory");
  }

  // Overlap with the async copy: load the 16x256 f32 A tile and convert to
  // 8 bf16 fragments. 16-bit A 16x32 layout: elem j -> K = j + 8*(j>>3) + 8*kh.
  v16bf afrag[8];
  const float* __restrict__ row = value + (size_t)(b * T_ + t0 + m) * D_;
#pragma unroll
  for (int kb = 0; kb < 8; ++kb) {
    const float* p = row + kb * 32 + kh * 8;   // 32B-aligned runs of 8 floats
#pragma unroll
    for (int j = 0; j < 8; ++j) afrag[kb][j] = (__bf16)p[j];
#pragma unroll
    for (int j = 0; j < 8; ++j) afrag[kb][8 + j] = (__bf16)p[16 + j];
  }

  asm volatile("s_wait_asynccnt 0" ::: "memory");
  __syncthreads();

  const v16bf* __restrict__ sBv = (const v16bf*)sB;

  float sacc[8];
#pragma unroll
  for (int r = 0; r < 8; ++r) sacc[r] = 0.f;

#pragma unroll 1
  for (int nb = 0; nb < 16; ++nb) {
    // Two independent accumulator chains over K for ILP.
    v8f c0 = {}, c1 = {};
#pragma unroll
    for (int kb = 0; kb < 4; ++kb) {
      const v16bf bfrag = sBv[(nb * 8 + kb) * 32 + lane];
      c0 = __builtin_amdgcn_wmma_f32_16x16x32_bf16(
               false, afrag[kb], false, bfrag, (short)0, c0, false, false);
    }
#pragma unroll
    for (int kb = 4; kb < 8; ++kb) {
      const v16bf bfrag = sBv[(nb * 8 + kb) * 32 + lane];
      c1 = __builtin_amdgcn_wmma_f32_16x16x32_bf16(
               false, afrag[kb], false, bfrag, (short)0, c1, false, false);
    }
    // C/D layout: VGPR r, lane l -> M = r + 8*kh, N = lane&15.
    const float wq = w1q[b * U_ + nb * 16 + m];
    const float sc = scale[nb * 16 + m];
#pragma unroll
    for (int r = 0; r < 8; ++r)
      sacc[r] += sc * fast_tanh(c0[r] + c1[r] + wq);
  }

  // Reduce over the 16 N-lanes within each kh half (xor 1,2,4,8 keeps bit4).
#pragma unroll
  for (int mask = 1; mask <= 8; mask <<= 1) {
#pragma unroll
    for (int r = 0; r < 8; ++r)
      sacc[r] += __shfl_xor(sacc[r], mask, 32);
  }
  if (m == 0) {
#pragma unroll
    for (int r = 0; r < 8; ++r)
      scores[(size_t)b * T_ + t0 + kh * 8 + r] = sacc[r];
  }
}

// ---------------------------------------------------------------------------
// Kernel 2: softmax over T per batch row.
// ---------------------------------------------------------------------------
__global__ __launch_bounds__(256) void softmax_kernel(
    const float* __restrict__ scores, float* __restrict__ attn) {
  __shared__ float red[256];
  const int b = blockIdx.x, tid = threadIdx.x;
  float loc[8];
  float mx = -3.4e38f;
#pragma unroll
  for (int i = 0; i < 8; ++i) {
    loc[i] = scores[(size_t)b * T_ + i * 256 + tid];
    mx = fmaxf(mx, loc[i]);
  }
  red[tid] = mx; __syncthreads();
  for (int s = 128; s > 0; s >>= 1) {
    if (tid < s) red[tid] = fmaxf(red[tid], red[tid + s]);
    __syncthreads();
  }
  mx = red[0]; __syncthreads();

  float sum = 0.f;
#pragma unroll
  for (int i = 0; i < 8; ++i) { loc[i] = __expf(loc[i] - mx); sum += loc[i]; }
  red[tid] = sum; __syncthreads();
  for (int s = 128; s > 0; s >>= 1) {
    if (tid < s) red[tid] += red[tid + s];
    __syncthreads();
  }
  const float inv = 1.f / red[0];
#pragma unroll
  for (int i = 0; i < 8; ++i)
    attn[(size_t)b * T_ + i * 256 + tid] = loc[i] * inv;
}

// ---------------------------------------------------------------------------
// Kernel 3: context[b][d] = sum_t attn[b][t] * value[b][t][d]
// Second pass over value hits L2 (128 MiB fits in 192 MB).
// ---------------------------------------------------------------------------
__global__ __launch_bounds__(256) void context_kernel(
    const float* __restrict__ value, const float* __restrict__ attn,
    float* __restrict__ ctx) {
  __shared__ float part[256][4];
  const int b = blockIdx.x, tid = threadIdx.x;
  const int dg = tid & 63;    // float4 group within D (64 * 4 = 256)
  const int tr = tid >> 6;    // 4-way split over T
  const float4* __restrict__ v4 = (const float4*)(value + (size_t)b * T_ * D_);
  const float* __restrict__ a = attn + (size_t)b * T_;
  float ax = 0.f, ay = 0.f, az = 0.f, aw = 0.f;
  for (int t = tr; t < T_; t += 4) {
    const float w = a[t];
    const float4 v = v4[t * 64 + dg];
    ax = fmaf(w, v.x, ax); ay = fmaf(w, v.y, ay);
    az = fmaf(w, v.z, az); aw = fmaf(w, v.w, aw);
  }
  part[tid][0] = ax; part[tid][1] = ay; part[tid][2] = az; part[tid][3] = aw;
  __syncthreads();
  if (tr == 0) {
    float4 o;
    o.x = part[dg][0] + part[dg + 64][0] + part[dg + 128][0] + part[dg + 192][0];
    o.y = part[dg][1] + part[dg + 64][1] + part[dg + 128][1] + part[dg + 192][1];
    o.z = part[dg][2] + part[dg + 64][2] + part[dg + 128][2] + part[dg + 192][2];
    o.w = part[dg][3] + part[dg + 64][3] + part[dg + 128][3] + part[dg + 192][3];
    ((float4*)ctx)[b * 64 + dg] = o;
  }
}

// ---------------------------------------------------------------------------
extern "C" void kernel_launch(void* const* d_in, const int* in_sizes, int n_in,
                              void* d_out, int out_size, void* d_ws, size_t ws_size,
                              hipStream_t stream) {
  const float* value = (const float*)d_in[0];  // [B, T, D]
  const float* query = (const float*)d_in[1];  // [B, D]
  const float* W1    = (const float*)d_in[2];  // [D, U]
  const float* W2    = (const float*)d_in[3];  // [D, U]
  const float* scale = (const float*)d_in[4];  // [U]

  float* ctx_out  = (float*)d_out;             // context [B,1,D] = 16384 floats
  float* attn_out = ctx_out + B_ * D_;         // attn    [B,1,T] = 131072 floats

  char*   ws     = (char*)d_ws;
  float*  w1q    = (float*)ws;                       // 64 KiB
  __bf16* packB  = (__bf16*)(ws + 65536);            // 128 KiB
  float*  scores = (float*)(ws + 65536 + 131072);    // 512 KiB

  w1q_kernel<<<B_, U_, 0, stream>>>(query, W1, w1q);
  packB_kernel<<<128, 32, 0, stream>>>(W2, packB);
  scores_kernel<<<dim3(B_, 16), 256, 0, stream>>>(value, w1q, scale, packB, scores);
  softmax_kernel<<<B_, 256, 0, stream>>>(scores, attn_out);
  context_kernel<<<B_, 256, 0, stream>>>(value, attn_out, ctx_out);
}
